// CustomRNNmodel_78159814853000
// MI455X (gfx1250) — matrix operand order repaired and checked by reading, not compiled, not run on hardware
//
#include <hip/hip_runtime.h>
#include <stdint.h>

// Problem constants (match reference)
#define Vv 32000
#define Hh 1024
#define Ll 2
#define Bb 4
#define Tt 1024
#define EPSc 1e-5f
#define NBLK 64          // persistent LSTM workgroups
#define CPB  16          // hidden columns owned per block = Hh/NBLK

// GEMM register blocking: each wave owns a 64x32 output patch
#define GEMM_MT 4        // 16-row tiles per wave
#define GEMM_NT 2        // 16-col tiles per wave

typedef __attribute__((ext_vector_type(16))) __bf16        v16bf;
typedef __attribute__((ext_vector_type(8)))  float         v8f;
typedef __attribute__((ext_vector_type(8)))  unsigned short ushort8;
typedef __attribute__((ext_vector_type(16))) unsigned short ushort16;

__device__ __forceinline__ float bf2f(unsigned short u) {
  return __uint_as_float(((unsigned)u) << 16);
}
__device__ __forceinline__ unsigned short f2bf(float f) {
  unsigned x = __float_as_uint(f);
  x += 0x7fffu + ((x >> 16) & 1u);     // round-to-nearest-even
  return (unsigned short)(x >> 16);
}
__device__ __forceinline__ float sigm(float x) { return 1.f / (1.f + __expf(-x)); }

// ---------------------------------------------------------------------------
// fp32 -> bf16 bulk convert (weights, vocab): done once, keeps L2 footprint low
// ---------------------------------------------------------------------------
__global__ void f32_to_bf16_kernel(const float* __restrict__ in,
                                   unsigned short* __restrict__ out, size_t n) {
  size_t i = (size_t)blockIdx.x * blockDim.x + threadIdx.x;
  size_t stride = (size_t)gridDim.x * blockDim.x;
  for (; i < n; i += stride) out[i] = f2bf(in[i]);
}

// ---------------------------------------------------------------------------
// Embedding: feats[b,t,:] = word_emb[id] + pos_emb[t]; emit fp32 + bf16 copy
// ---------------------------------------------------------------------------
__global__ __launch_bounds__(256) void embed_kernel(
    const int* __restrict__ ids, const float* __restrict__ wemb,
    const float* __restrict__ pemb, float* __restrict__ feats,
    unsigned short* __restrict__ featsb) {
  const int r = blockIdx.x;          // r = b*T + t
  const int t = r & (Tt - 1);
  const int id = ids[r];
  const float* w = wemb + (size_t)id * Hh;
  const float* p = pemb + (size_t)t * Hh;
  for (int h = threadIdx.x; h < Hh; h += 256) {
    float v = w[h] + p[h];
    feats[(size_t)r * Hh + h] = v;
    featsb[(size_t)r * Hh + h] = f2bf(v);
  }
}

// ---------------------------------------------------------------------------
// bf16 WMMA GEMM: C[M,N] = A[M,K] @ Bw[N,K]^T (+ bias0[n] + bias1[n])
// Each wave computes a 64x32 patch (4 M-tiles x 2 N-tiles): every B fragment
// is reused 4x and every A fragment 2x -> 8 WMMAs per 12 b128 loads, cutting
// L2 fragment traffic ~3x vs one-tile-per-wave. 8 waves/block cover 64x256.
// Fragment layouts as loaded here (wave32; contiguous k, same order for A and B):
//   A 16x32 bf16 : lane m=lane&15, kbase=(lane>>4)*16; elems[0..7]=kbase+0..7,
//                  elems[8..15]=kbase+8..15
//   B 32x16 bf16 : lane n=lane&15, elems[e] = K=(lane>>4)*16+e  (contiguous in
//                  Bw row n since Bw is [N,K] row-major == B^T)
//   C/D f32      : vgpr r -> M = r + 8*(lane>>4), N = lane&15
// ---------------------------------------------------------------------------
__global__ __launch_bounds__(256) void gemm_bf16_wmma(
    const unsigned short* __restrict__ A,   // [M x K] bf16 row-major
    const unsigned short* __restrict__ Bw,  // [N x K] bf16 row-major
    const float* __restrict__ bias0, const float* __restrict__ bias1,
    float* __restrict__ C, int M, int N, int K) {
  const int lane = threadIdx.x & 31;
  const int wave = threadIdx.x >> 5;
  const int lo = lane & 15, hi = lane >> 4;
  const int n0 = (blockIdx.x * 8 + wave) * (16 * GEMM_NT);  // wave-uniform
  const int m0 = blockIdx.y * (16 * GEMM_MT);
  if (n0 + 16 * GEMM_NT > N || m0 + 16 * GEMM_MT > M) return;  // uniform guard

  const unsigned short* arow[GEMM_MT];
  const unsigned short* brow[GEMM_NT];
#pragma unroll
  for (int i = 0; i < GEMM_MT; ++i)
    arow[i] = A + (size_t)(m0 + i * 16 + lo) * K;
#pragma unroll
  for (int j = 0; j < GEMM_NT; ++j)
    brow[j] = Bw + (size_t)(n0 + j * 16 + lo) * K;

  v8f acc[GEMM_MT][GEMM_NT] = {};
  for (int k0 = 0; k0 < K; k0 += 32) {
    v16bf af[GEMM_MT], bfv[GEMM_NT];
#pragma unroll
    for (int j = 0; j < GEMM_NT; ++j) {
      __builtin_prefetch(brow[j] + k0 + 512, 0, 3);  // keep streaming B warm
      ushort16 ub = *reinterpret_cast<const ushort16*>(brow[j] + k0 + hi * 16);
      bfv[j] = __builtin_bit_cast(v16bf, ub);
    }
#pragma unroll
    for (int i = 0; i < GEMM_MT; ++i) {
      ushort8 a0 = *reinterpret_cast<const ushort8*>(arow[i] + k0 + hi * 16);
      ushort8 a1 = *reinterpret_cast<const ushort8*>(arow[i] + k0 + hi * 16 + 8);
      ushort16 ua;
#pragma unroll
      for (int e = 0; e < 8; ++e) { ua[e] = a0[e]; ua[e + 8] = a1[e]; }
      af[i] = __builtin_bit_cast(v16bf, ua);
    }
#pragma unroll
    for (int i = 0; i < GEMM_MT; ++i)
#pragma unroll
      for (int j = 0; j < GEMM_NT; ++j)
        acc[i][j] = __builtin_amdgcn_wmma_f32_16x16x32_bf16(
            false, af[i], false, bfv[j], (short)0, acc[i][j], false, false);
  }

#pragma unroll
  for (int j = 0; j < GEMM_NT; ++j) {
    float bsum = 0.f;
    if (bias0) bsum += bias0[n0 + j * 16 + lo];
    if (bias1) bsum += bias1[n0 + j * 16 + lo];
#pragma unroll
    for (int i = 0; i < GEMM_MT; ++i) {
      float* crow = C + (size_t)(m0 + i * 16 + hi * 8) * N + (n0 + j * 16 + lo);
#pragma unroll
      for (int r = 0; r < 8; ++r) crow[(size_t)r * N] = acc[i][j][r] + bsum;
    }
  }
}

// ---------------------------------------------------------------------------
// Grid-wide barrier for the persistent LSTM (monotone generation counter)
// ---------------------------------------------------------------------------
__device__ __forceinline__ void grid_sync(unsigned* bar, unsigned target) {
  __threadfence();                 // release our stores to device scope
  __syncthreads();
  if (threadIdx.x == 0) {
    atomicAdd(bar, 1u);
    while (__hip_atomic_load(bar, __ATOMIC_ACQUIRE, __HIP_MEMORY_SCOPE_AGENT) <
           target)
      __builtin_amdgcn_s_sleep(1);
  }
  __syncthreads();
  __threadfence();                 // acquire: invalidate per-CU caches
}

__global__ void init_kernel(float* __restrict__ h0g, float* __restrict__ h1g,
                            unsigned* __restrict__ bar) {
  if (threadIdx.x == 0) *bar = 0u;
  for (int i = threadIdx.x; i < Bb * Hh; i += 256) { h0g[i] = 0.f; h1g[i] = 0.f; }
}

// ---------------------------------------------------------------------------
// Persistent 2-layer LSTM. 64 blocks x 256 threads; block owns 16 h-columns.
// Thread tid -> (b = tid>>6, gate g = (tid>>4)&3, col ci = tid&15): exactly one
// length-1024 dot per thread per matvec. Layer-0 input gates (feats@W_ih0^T +
// biases) were precomputed by the WMMA GEMM into Xg0. Weights are bf16
// (L2-resident, ~25 MB/step traffic); h state exchanged via global (L2) with
// two device-scope barriers per timestep.
// ---------------------------------------------------------------------------
__global__ __launch_bounds__(256) void lstm_persistent(
    const float* __restrict__ Xg0,            // [B*T, 4H]
    const unsigned short* __restrict__ Wih,   // [L,4H,H] bf16
    const unsigned short* __restrict__ Whh,   // [L,4H,H] bf16
    const float* __restrict__ bih, const float* __restrict__ bhh,
    float* __restrict__ h0g, float* __restrict__ h1g,
    float* __restrict__ outx,                 // [B,T,H]
    unsigned* __restrict__ bar) {
  __shared__ float h0s[Bb][Hh];
  __shared__ float h1s[Bb][Hh];
  __shared__ float gat[256];

  const int tid = threadIdx.x;
  const int cb = blockIdx.x * CPB;
  const int b  = tid >> 6;
  const int g  = (tid >> 4) & 3;
  const int ci = tid & 15;
  const int col = cb + ci;
  const int n   = g * Hh + col;               // gate-row index (i,f,g,o order)

  const int b2 = tid >> 4;                    // state-update threads (tid<64)
  const int col2 = cb + (tid & 15);
  float c0v = 0.f, c1v = 0.f;

  const unsigned short* Whh0 = Whh;
  const unsigned short* Wih1 = Wih + (size_t)4 * Hh * Hh;
  const unsigned short* Whh1 = Whh + (size_t)4 * Hh * Hh;

  unsigned gen = 0;
  for (int t = 0; t < Tt; ++t) {
    // stage h_prev (both layers) into LDS
    for (int i = tid; i < Bb * Hh; i += 256) {
      (&h0s[0][0])[i] = h0g[i];
      (&h1s[0][0])[i] = h1g[i];
    }
    __syncthreads();

    // ---- layer 0: gates = Xg0 + h0 . W_hh0_row ----
    {
      const unsigned short* wp = Whh0 + (size_t)n * Hh;
      float acc = Xg0[((size_t)b * Tt + t) * (4 * Hh) + n];
      for (int k = 0; k < Hh; k += 8) {
        ushort8 w = *reinterpret_cast<const ushort8*>(wp + k);
#pragma unroll
        for (int j = 0; j < 8; ++j) acc += h0s[b][k + j] * bf2f(w[j]);
      }
      gat[tid] = acc;
    }
    __syncthreads();
    if (tid < 64) {
      const int c = tid & 15;
      float iv = sigm(gat[b2 * 64 + c]);
      float fv = sigm(gat[b2 * 64 + 16 + c]);
      float gv = tanhf(gat[b2 * 64 + 32 + c]);
      float ov = sigm(gat[b2 * 64 + 48 + c]);
      c0v = fv * c0v + iv * gv;
      h0g[b2 * Hh + col2] = ov * tanhf(c0v);
    }
    ++gen; grid_sync(bar, gen * NBLK);

    // reload current-step h0 (written by all blocks)
    for (int i = tid; i < Bb * Hh; i += 256) (&h0s[0][0])[i] = h0g[i];
    __syncthreads();

    // ---- layer 1: gates = b + h0 . W_ih1_row + h1 . W_hh1_row ----
    {
      const unsigned short* wi = Wih1 + (size_t)n * Hh;
      const unsigned short* wh = Whh1 + (size_t)n * Hh;
      float acc = bih[4 * Hh + n] + bhh[4 * Hh + n];
      for (int k = 0; k < Hh; k += 8) {
        ushort8 a  = *reinterpret_cast<const ushort8*>(wi + k);
        ushort8 bw = *reinterpret_cast<const ushort8*>(wh + k);
#pragma unroll
        for (int j = 0; j < 8; ++j)
          acc += h0s[b][k + j] * bf2f(a[j]) + h1s[b][k + j] * bf2f(bw[j]);
      }
      gat[tid] = acc;
    }
    __syncthreads();
    if (tid < 64) {
      const int c = tid & 15;
      float iv = sigm(gat[b2 * 64 + c]);
      float fv = sigm(gat[b2 * 64 + 16 + c]);
      float gv = tanhf(gat[b2 * 64 + 32 + c]);
      float ov = sigm(gat[b2 * 64 + 48 + c]);
      c1v = fv * c1v + iv * gv;
      float hv = ov * tanhf(c1v);
      h1g[b2 * Hh + col2] = hv;
      outx[((size_t)b2 * Tt + t) * Hh + col2] = hv;
    }
    ++gen; grid_sync(bar, gen * NBLK);
  }
}

// ---------------------------------------------------------------------------
// LayerNorm over H, emit bf16 rows for the WMMA LM head
// ---------------------------------------------------------------------------
__global__ __launch_bounds__(256) void ln_kernel(
    const float* __restrict__ x, const float* __restrict__ w,
    const float* __restrict__ bsh, unsigned short* __restrict__ y) {
  __shared__ float r1[256], r2[256];
  const int row = blockIdx.x;
  const float* xr = x + (size_t)row * Hh;
  float v[4], s = 0.f, s2 = 0.f;
#pragma unroll
  for (int j = 0; j < 4; ++j) {
    v[j] = xr[threadIdx.x + j * 256];
    s += v[j]; s2 += v[j] * v[j];
  }
  r1[threadIdx.x] = s; r2[threadIdx.x] = s2;
  __syncthreads();
  for (int off = 128; off > 0; off >>= 1) {
    if (threadIdx.x < off) {
      r1[threadIdx.x] += r1[threadIdx.x + off];
      r2[threadIdx.x] += r2[threadIdx.x + off];
    }
    __syncthreads();
  }
  const float mu = r1[0] * (1.f / Hh);
  const float var = r2[0] * (1.f / Hh) - mu * mu;
  const float rs = rsqrtf(var + EPSc);
#pragma unroll
  for (int j = 0; j < 4; ++j) {
    int h = threadIdx.x + j * 256;
    y[(size_t)row * Hh + h] = f2bf((v[j] - mu) * rs * w[h] + bsh[h]);
  }
}

// ---------------------------------------------------------------------------
extern "C" void kernel_launch(void* const* d_in, const int* in_sizes, int n_in,
                              void* d_out, int out_size, void* d_ws,
                              size_t ws_size, hipStream_t stream) {
  (void)in_sizes; (void)n_in; (void)out_size; (void)ws_size;
  const int*   ids  = (const int*)d_in[0];
  const float* wemb = (const float*)d_in[1];
  const float* pemb = (const float*)d_in[2];
  const float* Wih  = (const float*)d_in[3];
  const float* Whh  = (const float*)d_in[4];
  const float* bih  = (const float*)d_in[5];
  const float* bhh  = (const float*)d_in[6];
  const float* lnw  = (const float*)d_in[7];
  const float* lnb  = (const float*)d_in[8];
  float* logits = (float*)d_out;

  char* wsp = (char*)d_ws;
  auto alloc = [&](size_t bytes) -> char* {
    char* p = wsp;
    wsp += (bytes + 255) & ~(size_t)255;
    return p;
  };
  unsigned short* WihB   = (unsigned short*)alloc((size_t)Ll * 4 * Hh * Hh * 2);
  unsigned short* WhhB   = (unsigned short*)alloc((size_t)Ll * 4 * Hh * Hh * 2);
  unsigned short* VocB   = (unsigned short*)alloc((size_t)Vv * Hh * 2);
  float*          feats  = (float*)alloc((size_t)Bb * Tt * Hh * 4);
  unsigned short* featsB = (unsigned short*)alloc((size_t)Bb * Tt * Hh * 2);
  float*          Xg0    = (float*)alloc((size_t)Bb * Tt * 4 * Hh * 4);
  float*          outx   = (float*)alloc((size_t)Bb * Tt * Hh * 4);
  unsigned short* xlnB   = (unsigned short*)alloc((size_t)Bb * Tt * Hh * 2);
  float*          h0g    = (float*)alloc((size_t)Bb * Hh * 4);
  float*          h1g    = (float*)alloc((size_t)Bb * Hh * 4);
  unsigned*       bar    = (unsigned*)alloc(256);

  // 1) bf16 conversions (weights + tied vocab)
  f32_to_bf16_kernel<<<2048, 256, 0, stream>>>(Wih, WihB, (size_t)Ll * 4 * Hh * Hh);
  f32_to_bf16_kernel<<<2048, 256, 0, stream>>>(Whh, WhhB, (size_t)Ll * 4 * Hh * Hh);
  f32_to_bf16_kernel<<<4096, 256, 0, stream>>>(wemb, VocB, (size_t)Vv * Hh);

  // 2) embeddings
  embed_kernel<<<Bb * Tt, 256, 0, stream>>>(ids, wemb, pemb, feats, featsB);

  // 3) hoisted layer-0 input gates: Xg0 = feats @ W_ih0^T + b_ih0 + b_hh0 (WMMA)
  //    per-block coverage: 64 rows x 256 cols
  gemm_bf16_wmma<<<dim3((4 * Hh) / 256, (Bb * Tt) / 64), 256, 0, stream>>>(
      featsB, WihB, bih, bhh, Xg0, Bb * Tt, 4 * Hh, Hh);

  // 4) persistent recurrent part
  init_kernel<<<1, 256, 0, stream>>>(h0g, h1g, bar);
  lstm_persistent<<<NBLK, 256, 0, stream>>>(Xg0, WihB, WhhB, bih, bhh, h0g, h1g,
                                            outx, bar);

  // 5) LayerNorm -> bf16
  ln_kernel<<<Bb * Tt, 256, 0, stream>>>(outx, lnw, lnb, xlnB);

  // 6) tied LM head: logits = xln @ word_emb^T (WMMA, HBM-write bound)
  gemm_bf16_wmma<<<dim3(Vv / 256, (Bb * Tt) / 64), 256, 0, stream>>>(
      xlnB, VocB, nullptr, nullptr, logits, Bb * Tt, Vv, Hh);
}
